// Attention_51110110822759
// MI455X (gfx1250) — matrix operand order repaired and checked by reading, not compiled, not run on hardware
//
#include <hip/hip_runtime.h>

// ---------------------------------------------------------------------------
// MI455X (gfx1250) fused QKV + causal flash-attention, f16 WMMA / f32 accum.
//   Kernel 1: qkv_proj  — x[f32] @ {Wq,Wk,Wv} -> Q,K,V stored f16 in d_ws
//   Kernel 2: flash_attn — online-softmax causal attention with
//             v_wmma_f32_16x16x32_f16 for Q·K^T and P·V; K tile staged with
//             GLOBAL_LOAD_ASYNC_TO_LDS_B128 (ASYNCcnt) when available.
// ---------------------------------------------------------------------------

typedef __attribute__((ext_vector_type(16))) _Float16 v16h;
typedef __attribute__((ext_vector_type(8)))  _Float16 v8h;
typedef __attribute__((ext_vector_type(8)))  float    v8f;
typedef __attribute__((ext_vector_type(4)))  float    v4f;
typedef __attribute__((ext_vector_type(4)))  int      v4i;

#define BATCH  16
#define SEQ    2048
#define DMODEL 1024
#define HDIM   64
#define BSROWS (BATCH * SEQ)

#if __has_builtin(__builtin_amdgcn_global_load_async_to_lds_b128)
#define USE_ASYNC_LDS 1
#define GLOBAL_AS __attribute__((address_space(1)))
#define LDS_AS    __attribute__((address_space(3)))
#else
#define USE_ASYNC_LDS 0
#endif

static __device__ inline void wait_async_zero() {
#if USE_ASYNC_LDS
#if __has_builtin(__builtin_amdgcn_s_wait_asynccnt)
  __builtin_amdgcn_s_wait_asynccnt(0);
#else
  asm volatile("s_wait_asynccnt 0x0" ::: "memory");
#endif
#endif
}

// ---- fragment loaders (wave32, 16x16x32 f16 WMMA) -------------------------
// A (16x32, row per lane): elem e -> K = (e/8)*16 + (lane/16)*8 + e%8
static __device__ inline v16h frag_a_f16(const _Float16* row, int lane) {
  const int off = ((lane >> 4) & 1) * 8;
  v8h lo = *(const v8h*)(row + off);
  v8h hi = *(const v8h*)(row + 16 + off);
  v16h r;
#pragma unroll
  for (int i = 0; i < 8; ++i) { r[i] = lo[i]; r[i + 8] = hi[i]; }
  return r;
}
// B (32x16, N per lane, K contiguous): elem e -> K = (e/8)*16 + (lane/16)*8 + e%8
// `row` points at the 32-long K-vector belonging to this lane's N column.
static __device__ inline v16h frag_b_f16(const _Float16* row, int lane) {
  const int off = ((lane >> 4) & 1) * 8;
  v8h lo = *(const v8h*)(row + off);
  v8h hi = *(const v8h*)(row + 16 + off);
  v16h r;
#pragma unroll
  for (int i = 0; i < 8; ++i) { r[i] = lo[i]; r[i + 8] = hi[i]; }
  return r;
}

static __device__ inline v8f wmma_f16(v16h a, v16h b, v8f c) {
  return __builtin_amdgcn_wmma_f32_16x16x32_f16(false, a, false, b,
                                                (short)0, c, false, false);
}

// ---------------------------------------------------------------------------
// Kernel 1: QKV projection.  grid = BSROWS/128, block = 256 (8 waves).
// Each wave: 16 rows x 64 cols for q,k,v (12 accumulators). x read once.
// ---------------------------------------------------------------------------
__global__ __launch_bounds__(256)
void qkv_proj_kernel(const float* __restrict__ x,
                     const float* __restrict__ Wq, const float* __restrict__ bq,
                     const float* __restrict__ Wk, const float* __restrict__ bk,
                     const float* __restrict__ Wv, const float* __restrict__ bv,
                     _Float16* __restrict__ Qo, _Float16* __restrict__ Ko,
                     _Float16* __restrict__ Vo)
{
  constexpr int WTP = 40;                      // padded K-stride (f16), 80B
  __shared__ _Float16 Wt[3 * 64 * WTP];        // W^T tiles: [m][n][k], 15 KB

  const int tid  = threadIdx.x;
  const int w    = tid >> 5;
  const int lane = tid & 31;
  const int hh   = (lane >> 4) & 1;
  const int l16  = lane & 15;
  const int row0 = blockIdx.x * 128 + w * 16;

  const float* Wm[3] = {Wq, Wk, Wv};
  const float* bm[3] = {bq, bk, bv};
  _Float16*    Om[3] = {Qo, Ko, Vo};

  v8f acc[3][4];
#pragma unroll
  for (int m = 0; m < 3; ++m)
#pragma unroll
    for (int nt = 0; nt < 4; ++nt) acc[m][nt] = {};

  for (int kk = 0; kk < DMODEL; kk += 32) {
    // cooperative stage: W[kk..kk+31][0..63] -> Wt[m][n][k] as f16
    for (int idx = tid; idx < 3 * 32 * 64; idx += 256) {
      const int m   = idx >> 11;
      const int rem = idx & 2047;
      const int k   = rem >> 6;
      const int n   = rem & 63;
      Wt[m * 64 * WTP + n * WTP + k] = (_Float16)Wm[m][(kk + k) * HDIM + n];
    }
    __syncthreads();

    // A fragment from x (f32 -> f16), 4x dwordx4 loads + pack
    const float* xr = x + (size_t)(row0 + l16) * DMODEL + kk;
    const int off = hh * 8;
    v4f a0 = *(const v4f*)(xr + off);
    v4f a1 = *(const v4f*)(xr + off + 4);
    v4f a2 = *(const v4f*)(xr + 16 + off);
    v4f a3 = *(const v4f*)(xr + 16 + off + 4);
    v16h afrag;
#pragma unroll
    for (int i = 0; i < 4; ++i) {
      afrag[i]      = (_Float16)a0[i];
      afrag[4 + i]  = (_Float16)a1[i];
      afrag[8 + i]  = (_Float16)a2[i];
      afrag[12 + i] = (_Float16)a3[i];
    }

    // batch 4 B-fragment loads per matrix, then 4 WMMAs (overlap DS latency)
#pragma unroll
    for (int m = 0; m < 3; ++m) {
      const _Float16* wb = &Wt[m * 64 * WTP];
      v16h bf0 = frag_b_f16(wb + (0 * 16 + l16) * WTP, lane);
      v16h bf1 = frag_b_f16(wb + (1 * 16 + l16) * WTP, lane);
      v16h bf2 = frag_b_f16(wb + (2 * 16 + l16) * WTP, lane);
      v16h bf3 = frag_b_f16(wb + (3 * 16 + l16) * WTP, lane);
      acc[m][0] = wmma_f16(afrag, bf0, acc[m][0]);
      acc[m][1] = wmma_f16(afrag, bf1, acc[m][1]);
      acc[m][2] = wmma_f16(afrag, bf2, acc[m][2]);
      acc[m][3] = wmma_f16(afrag, bf3, acc[m][3]);
    }
    __syncthreads();
  }

  // bias + store f16
#pragma unroll
  for (int m = 0; m < 3; ++m) {
#pragma unroll
    for (int nt = 0; nt < 4; ++nt) {
      const int col  = nt * 16 + l16;
      const float bb = bm[m][col];
#pragma unroll
      for (int r = 0; r < 8; ++r) {
        const int row = row0 + r + hh * 8;
        Om[m][(size_t)row * HDIM + col] = (_Float16)(acc[m][nt][r] + bb);
      }
    }
  }
}

// ---------------------------------------------------------------------------
// Kernel 2: causal flash attention.  grid = B*(S/64), block = 128 (4 waves).
// Wave = 16 query rows; WG shares 32-key K/V tiles staged in LDS.
// ---------------------------------------------------------------------------
__global__ __launch_bounds__(128)
void flash_attn_kernel(const _Float16* __restrict__ Qg,
                       const _Float16* __restrict__ Kg,
                       const _Float16* __restrict__ Vg,
                       float* __restrict__ Og)
{
  constexpr int KLP = 72;                   // K tile row stride (f16), 144B
  constexpr int VTP = 40;                   // V^T row stride, 80B
  constexpr int PLP = 40;                   // P row stride, 80B
  __shared__ _Float16 Kl[32 * KLP];         // keys x headdim
  __shared__ _Float16 Vt[64 * VTP];         // headdim x keys (transposed)
  __shared__ _Float16 Pl[4 * 16 * PLP];     // per-wave P tile (C -> A relayout)

  const int tid  = threadIdx.x;
  const int w    = tid >> 5;
  const int lane = tid & 31;
  const int hh   = (lane >> 4) & 1;
  const int l16  = lane & 15;

  const int b     = blockIdx.x >> 5;        // SEQ/64 = 32 q-tiles per batch
  const int qt    = blockIdx.x & 31;
  const int q0    = qt * 64;
  const int qrow0 = q0 + w * 16;
  const size_t base = (size_t)b * SEQ;

  // Q fragments for this wave's 16 rows (headdim 64 = 2 x K32)
  const _Float16* qr = Qg + (base + qrow0 + l16) * HDIM;
  v16h aQ0 = frag_a_f16(qr, lane);
  v16h aQ1 = frag_a_f16(qr + 32, lane);

  v8f accO[4];
#pragma unroll
  for (int nt = 0; nt < 4; ++nt) accO[nt] = {};
  float mrow[8], lrow[8];
#pragma unroll
  for (int r = 0; r < 8; ++r) { mrow[r] = -1e30f; lrow[r] = 0.f; }

  const int nkt = (q0 + 64) / 32;           // causal loop bound for this WG
  for (int kt = 0; kt < nkt; ++kt) {
    const int key0 = kt * 32;

    // ---- cooperative stage: K via async DMA to LDS, V transposed ---------
    for (int c = tid; c < 256; c += 128) {
      const int key = c >> 3;
      const int n8  = (c & 7) * 8;
#if USE_ASYNC_LDS
      __builtin_amdgcn_global_load_async_to_lds_b128(
          (GLOBAL_AS v4i*)(void*)(Kg + (base + key0 + key) * HDIM + n8),
          (LDS_AS v4i*)(void*)(&Kl[key * KLP + n8]),
          0, 0);
#else
      v8h kv = *(const v8h*)(Kg + (base + key0 + key) * HDIM + n8);
      *(v8h*)(&Kl[key * KLP + n8]) = kv;
#endif
      v8h vv = *(const v8h*)(Vg + (base + key0 + key) * HDIM + n8);
#pragma unroll
      for (int i = 0; i < 8; ++i) Vt[(n8 + i) * VTP + key] = vv[i];
      if (kt + 1 < nkt) {                   // warm L2 for next tile
        __builtin_prefetch(Kg + (base + key0 + 32 + key) * HDIM + n8, 0, 1);
        __builtin_prefetch(Vg + (base + key0 + 32 + key) * HDIM + n8, 0, 1);
      }
    }
    wait_async_zero();
    __syncthreads();

    // ---- S = Q K^T : two 16x16 C tiles covering 32 keys ------------------
    v16h b00 = frag_b_f16(&Kl[(0 * 16 + l16) * KLP + 0], lane);
    v16h b01 = frag_b_f16(&Kl[(0 * 16 + l16) * KLP + 32], lane);
    v16h b10 = frag_b_f16(&Kl[(1 * 16 + l16) * KLP + 0], lane);
    v16h b11 = frag_b_f16(&Kl[(1 * 16 + l16) * KLP + 32], lane);
    v8f sf[2] = {{}, {}};
    sf[0] = wmma_f16(aQ0, b00, sf[0]);
    sf[1] = wmma_f16(aQ0, b10, sf[1]);
    sf[0] = wmma_f16(aQ1, b01, sf[0]);
    sf[1] = wmma_f16(aQ1, b11, sf[1]);

    // ---- scale + causal mask (C layout: row = r + 8*hh, col = lane&15) ---
#pragma unroll
    for (int h = 0; h < 2; ++h) {
      const int key = key0 + h * 16 + l16;
#pragma unroll
      for (int r = 0; r < 8; ++r) {
        const int row = qrow0 + r + hh * 8;
        sf[h][r] = (key <= row) ? sf[h][r] * 0.125f : -1e30f;
      }
    }

    // ---- online softmax: per-row max via 16-lane butterfly ---------------
    float mnew[8];
#pragma unroll
    for (int r = 0; r < 8; ++r) {
      float t = fmaxf(sf[0][r], sf[1][r]);
#pragma unroll
      for (int mk = 1; mk < 16; mk <<= 1) t = fmaxf(t, __shfl_xor(t, mk, 32));
      mnew[r] = fmaxf(mrow[r], t);
    }
    float al[8], rs[8];
#pragma unroll
    for (int r = 0; r < 8; ++r) { al[r] = __expf(mrow[r] - mnew[r]); rs[r] = 0.f; }

    // P = exp(S - m); write f16 into per-wave LDS tile (C -> A relayout)
#pragma unroll
    for (int h = 0; h < 2; ++h) {
#pragma unroll
      for (int r = 0; r < 8; ++r) {
        float p = __expf(sf[h][r] - mnew[r]);
        rs[r] += p;
        Pl[w * 16 * PLP + (r + hh * 8) * PLP + h * 16 + l16] = (_Float16)p;
      }
    }
#pragma unroll
    for (int r = 0; r < 8; ++r) {
#pragma unroll
      for (int mk = 1; mk < 16; mk <<= 1) rs[r] += __shfl_xor(rs[r], mk, 32);
      lrow[r] = lrow[r] * al[r] + rs[r];
      mrow[r] = mnew[r];
    }
#pragma unroll
    for (int nt = 0; nt < 4; ++nt)
#pragma unroll
      for (int r = 0; r < 8; ++r) accO[nt][r] *= al[r];

    // ---- O += P * V  (same-wave DS store->load is in-order) --------------
    v16h pfrag = frag_a_f16(&Pl[w * 16 * PLP + l16 * PLP], lane);
    v16h vb0 = frag_b_f16(&Vt[(0 * 16 + l16) * VTP], lane);
    v16h vb1 = frag_b_f16(&Vt[(1 * 16 + l16) * VTP], lane);
    v16h vb2 = frag_b_f16(&Vt[(2 * 16 + l16) * VTP], lane);
    v16h vb3 = frag_b_f16(&Vt[(3 * 16 + l16) * VTP], lane);
    accO[0] = wmma_f16(pfrag, vb0, accO[0]);
    accO[1] = wmma_f16(pfrag, vb1, accO[1]);
    accO[2] = wmma_f16(pfrag, vb2, accO[2]);
    accO[3] = wmma_f16(pfrag, vb3, accO[3]);

    __syncthreads();                        // protect Kl/Vt for next stage
  }

  // ---- normalize and store fp32 output -----------------------------------
#pragma unroll
  for (int nt = 0; nt < 4; ++nt) {
    const int col = nt * 16 + l16;
#pragma unroll
    for (int r = 0; r < 8; ++r) {
      const int row = qrow0 + r + hh * 8;
      Og[(base + row) * HDIM + col] = accO[nt][r] / lrow[r];
    }
  }
}

// ---------------------------------------------------------------------------
extern "C" void kernel_launch(void* const* d_in, const int* in_sizes, int n_in,
                              void* d_out, int out_size, void* d_ws, size_t ws_size,
                              hipStream_t stream)
{
  const float* x  = (const float*)d_in[0];
  const float* Wq = (const float*)d_in[1];
  const float* bq = (const float*)d_in[2];
  const float* Wk = (const float*)d_in[3];
  const float* bk = (const float*)d_in[4];
  const float* Wv = (const float*)d_in[5];
  const float* bv = (const float*)d_in[6];
  float* out = (float*)d_out;

  _Float16* Qh = (_Float16*)d_ws;                    // 3 * 32768*64 f16 = 12.6 MB
  _Float16* Kh = Qh + (size_t)BSROWS * HDIM;
  _Float16* Vh = Kh + (size_t)BSROWS * HDIM;

  qkv_proj_kernel<<<BSROWS / 128, 256, 0, stream>>>(x, Wq, bq, Wk, bk, Wv, bv,
                                                    Qh, Kh, Vh);
  flash_attn_kernel<<<BATCH * (SEQ / 64), 128, 0, stream>>>(Qh, Kh, Vh, out);
}
